// MultiHeadRegionalAttention2D_44719199486444
// MI455X (gfx1250) — matrix operand order repaired and checked by reading, not compiled, not run on hardware
//
#include <hip/hip_runtime.h>
#include <hip/hip_bf16.h>

// ---- CDNA5 WMMA fragment types (wave32) ----
typedef __attribute__((ext_vector_type(16))) _Float16 v16h;
typedef __attribute__((ext_vector_type(8)))  float    v8f;

#define C_IN   128
#define NHEAD  8
#define DH     32          // channels per head
#define HIM    192
#define WIM    192
#define HW     (HIM*WIM)
#define TILE   8           // output tile edge
#define HALO   10          // tile + 1-pixel halo
#define NPIX   100         // 10*10 halo pixels
#define NPAD   112         // padded to 7 WMMA n-tiles of 16
#define NT     7
#define KSTEPS 4           // 128 / 32
#define CSTR   136         // x-halo LDS stride (halves): 272B row, 16B aligned
#define MSTR   264         // q/k/v LDS stride (halves): 528B row, 16B aligned

// LDS element-offset map (all arrays alias the same dynamic-LDS base)
#define LX_OFF 0                       // [NPAD][CSTR] f16 x-halo
#define A_OFF  (NPAD*CSTR)             // [NPAD][MSTR] f16 q (later v)
#define B_OFF  (A_OFF + NPAD*MSTR)     // [NPAD][MSTR] f16 k
#define S_OFFF ((B_OFF + NPAD*MSTR)/2) // [NHEAD][NPAD] f32 scores (float index)
#define LDS_BYTES ((B_OFF + NPAD*MSTR)*2 + NHEAD*NPAD*4)

union HFrag { v16h v; _Float16 h[16]; };

// Typed views of dynamic LDS — stay in addrspace(3), no generic casts.
extern __shared__ _Float16 ldsH[];
extern __shared__ float    ldsF[];

__global__ __launch_bounds__(256)
void mhra2d_fused_kernel(const float* __restrict__ x,
                         const float* __restrict__ wq,
                         const float* __restrict__ wk,
                         const float* __restrict__ wv,
                         float* __restrict__ out) {
    const int tid  = threadIdx.x;
    const int lane = tid & 31;
    const int wave = tid >> 5;
    const int x0 = blockIdx.x * TILE;
    const int y0 = blockIdx.y * TILE;

    const int m  = lane & 15;
    const int hi = lane >> 4;

    // Load the 4 B fragments (32x16 f16) of one N-tile from the LDS x-halo.
    auto load_b4 = [&](int nt, v16h* Bf) {
        const int p = nt * 16 + m;
#pragma unroll
        for (int ks = 0; ks < KSTEPS; ++ks) {
            HFrag b;
            const int off = LX_OFF + p * CSTR + ks * 32 + 8 * hi;
#pragma unroll
            for (int e = 0; e < 16; ++e) b.h[e] = ldsH[off + e + (e & 8)];
            Bf[ks] = b.v;
        }
    };

    // -----------------------------------------------------------------
    // Projection GEMM: ldsH[dstOff..] = Wmat[256x128] * xhalo[128xNPAD]
    // Each wave owns 2 M-tiles; A-fragments register-resident across NT;
    // B-fragments double-buffered across N-tiles to hide LDS latency.
    // -----------------------------------------------------------------
    auto proj_gemm = [&](const float* __restrict__ Wmat, int dstOff) {
#pragma unroll
        for (int mi = 0; mi < 2; ++mi) {
            const int mt  = wave * 2 + mi;
            const int row = mt * 16 + m;

            // ---- Load A fragments for all 4 K-steps: 16 loads, one wait ----
            float4 t[KSTEPS][4];
#pragma unroll
            for (int ks = 0; ks < KSTEPS; ++ks) {
                const float* base = Wmat + row * C_IN + ks * 32 + 8 * hi;
                t[ks][0] = *(const float4*)(base + 0);
                t[ks][1] = *(const float4*)(base + 4);
                t[ks][2] = *(const float4*)(base + 16);
                t[ks][3] = *(const float4*)(base + 20);
            }
            v16h A[KSTEPS];
#pragma unroll
            for (int ks = 0; ks < KSTEPS; ++ks) {
                HFrag a;
#pragma unroll
                for (int q = 0; q < 2; ++q) {    // q=0 -> K[0..8), q=1 -> K[16..24)
#pragma unroll
                    for (int e = 0; e < 4; ++e) {
                        const float4 f4 = t[ks][q * 2 + e / 2];
                        a.h[q * 8 + e * 2 + 0] = (_Float16)((e & 1) ? f4.z : f4.x);
                        a.h[q * 8 + e * 2 + 1] = (_Float16)((e & 1) ? f4.w : f4.y);
                    }
                }
                A[ks] = a.v;
            }

            // ---- N-tile loop, fully unrolled, B double-buffered ----
            v16h Bbuf[2][KSTEPS];
            load_b4(0, Bbuf[0]);
#pragma unroll
            for (int nt = 0; nt < NT; ++nt) {
                // prefetch next tile's B fragments before this tile's WMMAs
                if (nt + 1 < NT) load_b4(nt + 1, Bbuf[(nt + 1) & 1]);
                __builtin_amdgcn_sched_barrier(0);  // keep prefetch above WMMAs

                v8f acc = {};
#pragma unroll
                for (int ks = 0; ks < KSTEPS; ++ks)
                    acc = __builtin_amdgcn_wmma_f32_16x16x32_f16(
                        false, A[ks], false, Bbuf[nt & 1][ks],
                        (short)0, acc, false, false);

                // C/D layout: VGPR r -> M = mt*16 + 8*hi + r, N = lane&15
                const int p = nt * 16 + m;
                const int o = dstOff + p * MSTR + mt * 16 + 8 * hi;
#pragma unroll
                for (int r = 0; r < 8; ++r) ldsH[o + r] = (_Float16)acc[r];
            }
        }
    };

    // ---- Phase 1: stage x halo tile -> LDS f16, [pixel][channel] ----
    for (int idx = tid; idx < C_IN * NPAD; idx += 256) {
        const int c = idx / NPAD;
        const int p = idx - c * NPAD;
        float val = 0.f;
        if (p < NPIX) {
            const int hr = p / HALO, hc = p - hr * HALO;
            const int gy = y0 - 1 + hr, gx = x0 - 1 + hc;
            if ((unsigned)gy < (unsigned)HIM && (unsigned)gx < (unsigned)WIM)
                val = x[c * HW + gy * WIM + gx];
        }
        ldsH[LX_OFF + p * CSTR + c] = (_Float16)val;
    }
    __syncthreads();

    // ---- Phase 2: q and k projections (WMMA) ----
    proj_gemm(wq, A_OFF);
    proj_gemm(wk, B_OFF);
    __syncthreads();

    // ---- Phase 3: per-pixel per-head score field S = dot(q, k) ----
    for (int u = tid; u < NHEAD * NPAD; u += 256) {
        const int h = u / NPAD, p = u - h * NPAD;
        const int qo = A_OFF + p * MSTR + h * DH;
        const int ko = B_OFF + p * MSTR + h * DH;
        float dot = 0.f;
#pragma unroll
        for (int c = 0; c < DH; ++c)
            dot += (float)ldsH[qo + c] * (float)ldsH[ko + c];
        ldsF[S_OFFF + h * NPAD + p] = dot;
    }
    __syncthreads();

    // ---- Phase 4: v projection overwrites q buffer ----
    proj_gemm(wv, A_OFF);
    __syncthreads();

    // ---- Phase 5: 3x3 softmax over neighbor scores, blend neighbor v ----
    for (int u = tid; u < NHEAD * TILE * TILE; u += 256) {
        const int h  = u >> 6;
        const int po = u & 63;
        const int ly = po >> 3, lxc = po & 7;
        const int hr = ly + 1, hc = lxc + 1;   // halo coords of center pixel

        int   pn[9];
        float e[9];
        float mx = -1e30f;
#pragma unroll
        for (int s = 0; s < 9; ++s) {
            const int di = s / 3 - 1, dj = s % 3 - 1;
            pn[s] = (hr + di) * HALO + (hc + dj);
            const float sv = ldsF[S_OFFF + h * NPAD + pn[s]] * (1.0f / 16.0f);
            e[s] = sv;
            mx = fmaxf(mx, sv);
        }
        float sum = 0.f;
#pragma unroll
        for (int s = 0; s < 9; ++s) { e[s] = __expf(e[s] - mx); sum += e[s]; }
        const float inv = 1.0f / sum;

        float o[DH];
#pragma unroll
        for (int c = 0; c < DH; ++c) o[c] = 0.f;
#pragma unroll
        for (int s = 0; s < 9; ++s) {
            const float w = e[s] * inv;
            const int vo = A_OFF + pn[s] * MSTR + h * DH;   // v overwrote q
#pragma unroll
            for (int c = 0; c < DH; ++c) o[c] += w * (float)ldsH[vo + c];
        }

        const int gy = y0 + ly, gx = x0 + lxc;
        float* ob = out + (h * DH) * HW + gy * WIM + gx;
#pragma unroll
        for (int c = 0; c < DH; ++c) ob[c * HW] = o[c];
    }
}

extern "C" void kernel_launch(void* const* d_in, const int* in_sizes, int n_in,
                              void* d_out, int out_size, void* d_ws, size_t ws_size,
                              hipStream_t stream) {
    const float* x  = (const float*)d_in[0];
    const float* wq = (const float*)d_in[1];
    const float* wk = (const float*)d_in[2];
    const float* wv = (const float*)d_in[3];
    float* out = (float*)d_out;

    dim3 grid(WIM / TILE, HIM / TILE);   // 24 x 24 = 576 workgroups
    dim3 block(256);                     // 8 waves (wave32)
    size_t shmem = LDS_BYTES;            // 152320 B -> 2 workgroups per 320KB WGP
    mhra2d_fused_kernel<<<grid, block, shmem, stream>>>(x, wq, wk, wv, out);
}
